// RNNModel_45561013076655
// MI455X (gfx1250) — matrix operand order repaired and checked by reading, not compiled, not run on hardware
//
#include <hip/hip_runtime.h>

#define B_  256
#define T_  200
#define D_  1250
#define H1_ 100
#define H2_ 20
#define O_  2

typedef __attribute__((ext_vector_type(16))) __bf16 v16bf;
typedef __attribute__((ext_vector_type(8)))  float  v8f;
typedef __attribute__((ext_vector_type(2)))  float  f2;

__device__ __forceinline__ __bf16 tobf(float f) { return (__bf16)f; }

// ---------------------------------------------------------------------------
// C[M,N] = A[M,K] @ W[N,K]^T + ba[N] + bb[N]      (bf16 WMMA, f32 accumulate)
// One wave owns one 16-row M-tile and ALL NT column tiles (A loaded once per
// K-chunk). Main K-loop is guard-free with float2 loads; 2-element K tail is
// handled with clamp-address + select (no divergent loads).
// ---------------------------------------------------------------------------
template<int NT>
__global__ void wmma_gemm_bias(const float* __restrict__ A,
                               const float* __restrict__ W,
                               const float* __restrict__ ba,
                               const float* __restrict__ bb,
                               float* __restrict__ C,
                               int M, int N, int K)
{
    const int lane = threadIdx.x & 31;
    const int wave = threadIdx.x >> 5;
    const int mt   = blockIdx.x * (blockDim.x >> 5) + wave;
    if (mt * 16 >= M) return;                   // wave-uniform

    const int half = lane >> 4;
    const int l15  = lane & 15;

    int am = mt * 16 + l15;                     // A row owned by this lane
    if (am >= M) am = M - 1;                    // junk rows never stored
    const float* arow = A + (size_t)am * K;

    const float* wrow[NT];                      // B col (= W row) per N-tile
    #pragma unroll
    for (int nt = 0; nt < NT; ++nt) {
        int n = nt * 16 + l15;                  // garbage cols never stored
        wrow[nt] = W + (size_t)(n < N ? n : 0) * K;
    }

    v8f acc[NT];
    #pragma unroll
    for (int nt = 0; nt < NT; ++nt)
        acc[nt] = (v8f){0.f,0.f,0.f,0.f,0.f,0.f,0.f,0.f};

    const int kfull = K >> 5;
    for (int kc = 0; kc < kfull; ++kc) {
        const int kb = kc << 5;
        // A 16x32 bf16: lanes 0-15 K=0..15, lanes 16-31 K=16..31
        v16bf a;
        {
            const f2* p0 = (const f2*)(arow + kb + half * 16);
            const f2* p1 = (const f2*)(arow + kb + 8 + half * 16);
            #pragma unroll
            for (int e = 0; e < 4; ++e) {
                f2 v0 = p0[e], v1 = p1[e];
                a[2*e]     = tobf(v0.x);  a[2*e + 1]     = tobf(v0.y);
                a[2*e + 8] = tobf(v1.x);  a[2*e + 9]     = tobf(v1.y);
            }
        }
        #pragma unroll
        for (int nt = 0; nt < NT; ++nt) {
            // B 32x16 bf16: lane = col, lanes 0-15 K=0..15, lanes 16-31 K=16..31
            v16bf b;
            const f2* pb = (const f2*)(wrow[nt] + kb + half * 16);
            #pragma unroll
            for (int e = 0; e < 8; ++e) {
                f2 v = pb[e];
                b[2*e] = tobf(v.x);  b[2*e + 1] = tobf(v.y);
            }
            acc[nt] = __builtin_amdgcn_wmma_f32_16x16x32_bf16(
                          false, a, false, b, (short)0, acc[nt], false, false);
        }
    }
    if ((kfull << 5) < K) {                     // K tail (clamp + select, no branches)
        const int kb = kfull << 5;
        v16bf a;
        #pragma unroll
        for (int e = 0; e < 8; ++e) {
            int k0 = kb + half * 16 + e;
            int k1 = k0 + 8;
            float v0 = arow[k0 < K ? k0 : 0];
            float v1 = arow[k1 < K ? k1 : 0];
            a[e]     = tobf(k0 < K ? v0 : 0.f);
            a[e + 8] = tobf(k1 < K ? v1 : 0.f);
        }
        #pragma unroll
        for (int nt = 0; nt < NT; ++nt) {
            v16bf b;
            #pragma unroll
            for (int e = 0; e < 16; ++e) {
                int k = kb + half * 16 + e;
                float v = wrow[nt][k < K ? k : 0];
                b[e] = tobf(k < K ? v : 0.f);
            }
            acc[nt] = __builtin_amdgcn_wmma_f32_16x16x32_bf16(
                          false, a, false, b, (short)0, acc[nt], false, false);
        }
    }
    #pragma unroll
    for (int nt = 0; nt < NT; ++nt) {
        int n = nt * 16 + l15;                  // D: row = half*8+r, col = lane&15
        if (n < N) {
            float bias = ba[n] + bb[n];
            #pragma unroll
            for (int r = 0; r < 8; ++r) {
                int row = mt * 16 + half * 8 + r;
                if (row < M) C[(size_t)row * N + n] = acc[nt][r] + bias;
            }
        }
    }
}

// ---------------------------------------------------------------------------
// Layer-1 recurrent scan. Block = 256 threads (8 waves) owns 16 batch rows.
// W_hh1 (zero-padded, row stride 132 to dodge bank conflicts) and bf16 h-state
// live in LDS. Per step: waves 0..6 compute one 16-wide N tile via 4 WMMAs.
// Epilogue computes all values unconditionally, then stores under ONE branch.
// ---------------------------------------------------------------------------
__global__ void rnn1_scan(const float* __restrict__ pre1,   // [B,T,H1]
                          const float* __restrict__ Whh,    // [H1,H1]
                          const float* __restrict__ mask1,  // [B,T,H1]
                          float* __restrict__ out1m)        // [B,T,H1]
{
    __shared__ __bf16 Wl[112 * 132];
    __shared__ __bf16 hb[16 * 132];

    const int tid  = threadIdx.x;
    const int lane = tid & 31;
    const int wave = tid >> 5;
    const int half = lane >> 4;
    const int l15  = lane & 15;
    const int b0   = blockIdx.x * 16;

    for (int i = tid; i < 112 * 132; i += blockDim.x) {
        int r = i / 132, k = i - r * 132;
        Wl[i] = tobf((r < H1_ && k < H1_) ? Whh[r * H1_ + k] : 0.f);
    }
    for (int i = tid; i < 16 * 132; i += blockDim.x) hb[i] = tobf(0.f);
    __syncthreads();

    const bool active = (wave < 7);             // wave-uniform
    const int  n      = wave * 16 + l15;
    const bool nok    = active && (n < H1_);
    const int  ncl    = n < H1_ ? n : H1_ - 1;  // clamped col for loads

    for (int t = 0; t < T_; ++t) {
        float vout[8];
        if (active) {
            // hoist global loads (clamped addr, unconditional) ahead of WMMAs
            float pre[8], mk[8];
            #pragma unroll
            for (int r = 0; r < 8; ++r) {
                size_t idx = ((size_t)(b0 + half * 8 + r) * T_ + t) * H1_ + ncl;
                pre[r] = pre1[idx];
                mk[r]  = mask1[idx];
            }
            v8f acc = (v8f){0.f,0.f,0.f,0.f,0.f,0.f,0.f,0.f};
            #pragma unroll
            for (int kc = 0; kc < 4; ++kc) {    // K = 100 -> 4 chunks, zero-padded
                const int kb = kc << 5;
                v16bf a, b;
                #pragma unroll
                for (int e = 0; e < 8; ++e) {
                    int k0 = kb + half * 16 + e;
                    a[e]     = hb[l15 * 132 + k0];
                    a[e + 8] = hb[l15 * 132 + k0 + 8];
                }
                #pragma unroll
                for (int e = 0; e < 16; ++e)
                    b[e] = Wl[n * 132 + kb + half * 16 + e];
                acc = __builtin_amdgcn_wmma_f32_16x16x32_bf16(
                          false, a, false, b, (short)0, acc, false, false);
            }
            // compute everything unconditionally -> mask loads stay hoisted
            float pm[8];
            #pragma unroll
            for (int r = 0; r < 8; ++r) {
                float v = pre[r] + acc[r];
                v = v > 0.f ? v : 0.f;
                vout[r] = v;
                pm[r]   = v * mk[r];
            }
            if (nok) {                          // single exec-mask toggle per step
                #pragma unroll
                for (int r = 0; r < 8; ++r) {
                    size_t idx = ((size_t)(b0 + half * 8 + r) * T_ + t) * H1_ + n;
                    out1m[idx] = pm[r];
                }
            }
        }
        __syncthreads();                        // all waves done reading hb
        if (nok) {
            #pragma unroll
            for (int r = 0; r < 8; ++r)
                hb[(half * 8 + r) * 132 + n] = tobf(vout[r]);
        }
        __syncthreads();                        // hb updated for next step
    }
}

// ---------------------------------------------------------------------------
// Layer-2 recurrent scan (H2=20) + final dropout + FC. Scalar f32 VALU.
// Block = 160 threads = 8 batch rows x 20 hidden units (5 waves exactly).
// ---------------------------------------------------------------------------
__global__ void rnn2_scan_fc(const float* __restrict__ pre2,   // [B,T,H2]
                             const float* __restrict__ Whh2,   // [H2,H2]
                             const float* __restrict__ mask2,  // [B,T,H2]
                             const float* __restrict__ Wfc,    // [O,H2]
                             const float* __restrict__ bfc,    // [O]
                             float* __restrict__ out)          // [B,O]
{
    __shared__ float Wl[H2_ * H2_];
    __shared__ float h [8][H2_];
    __shared__ float hm[8][H2_];

    const int tid = threadIdx.x;
    const int bl  = tid / H2_;                  // 0..7
    const int n   = tid - bl * H2_;             // 0..19
    const int b   = blockIdx.x * 8 + bl;

    for (int i = tid; i < H2_ * H2_; i += blockDim.x) Wl[i] = Whh2[i];
    h[bl][n] = 0.f;
    __syncthreads();

    for (int t = 0; t < T_; ++t) {
        float s = pre2[((size_t)b * T_ + t) * H2_ + n];
        #pragma unroll
        for (int k = 0; k < H2_; ++k) s += h[bl][k] * Wl[n * H2_ + k];
        s = s > 0.f ? s : 0.f;
        __syncthreads();
        h[bl][n] = s;
        __syncthreads();
    }
    hm[bl][n] = h[bl][n] * mask2[((size_t)b * T_ + (T_ - 1)) * H2_ + n];
    __syncthreads();
    if (n < O_) {
        float s = bfc[n];
        #pragma unroll
        for (int k = 0; k < H2_; ++k) s += hm[bl][k] * Wfc[n * H2_ + k];
        out[(size_t)b * O_ + n] = s;
    }
}

// ---------------------------------------------------------------------------
extern "C" void kernel_launch(void* const* d_in, const int* in_sizes, int n_in,
                              void* d_out, int out_size, void* d_ws, size_t ws_size,
                              hipStream_t stream)
{
    const float* x     = (const float*)d_in[0];
    const float* Wih1  = (const float*)d_in[1];
    const float* Whh1  = (const float*)d_in[2];
    const float* bih1  = (const float*)d_in[3];
    const float* bhh1  = (const float*)d_in[4];
    const float* Wih2  = (const float*)d_in[5];
    const float* Whh2  = (const float*)d_in[6];
    const float* bih2  = (const float*)d_in[7];
    const float* bhh2  = (const float*)d_in[8];
    const float* Wfc   = (const float*)d_in[9];
    const float* bfcp  = (const float*)d_in[10];
    const float* mask1 = (const float*)d_in[11];
    const float* mask2 = (const float*)d_in[12];

    float* ws    = (float*)d_ws;
    float* pre1  = ws;                                   // 51200*100 f32
    float* out1m = pre1  + (size_t)B_ * T_ * H1_;        // 51200*100 f32
    float* pre2  = out1m + (size_t)B_ * T_ * H1_;        // 51200*20  f32
    float* outp  = (float*)d_out;                        // [256,2]

    const int M = B_ * T_;                               // 51200

    // Layer-1 input projection: 3200 M-tile waves, 7 N-tiles each in registers
    {
        int blocks = (M / 16 + 7) / 8;                   // 400
        wmma_gemm_bias<7><<<blocks, 256, 0, stream>>>(x, Wih1, bih1, bhh1,
                                                      pre1, M, H1_, D_);
    }
    // Layer-1 recurrence: 16 blocks x 16 batch rows
    rnn1_scan<<<16, 256, 0, stream>>>(pre1, Whh1, mask1, out1m);

    // Layer-2 input projection: 3200 M-tile waves, 2 N-tiles each
    {
        int blocks = (M / 16 + 7) / 8;                   // 400
        wmma_gemm_bias<2><<<blocks, 256, 0, stream>>>(out1m, Wih2, bih2, bhh2,
                                                      pre2, M, H2_, H1_);
    }
    // Layer-2 recurrence + FC: 32 blocks x 8 batch rows
    rnn2_scan_fc<<<32, 160, 0, stream>>>(pre2, Whh2, mask2, Wfc, bfcp, outp);
}
